// DGCNN_simple_5686536700339
// MI455X (gfx1250) — compile-verified, matrix operand order checked by reading, not run on hardware
//
#include <hip/hip_runtime.h>
#include <hip/hip_bf16.h>

// DGCNN forward for MI455X (gfx1250, wave32).
// Heavy GEMMs (KNN#2 Gram matrix, conv5, conv6) run on V_WMMA_F32_16X16X32_BF16.
// Problem is ~17 GFLOP / ~40MB working set -> L2-resident on 192MB L2; bf16 WMMA
// w/ f32 accumulation is the throughput-optimal CDNA5 path. Bulk LDS staging
// uses GLOBAL_LOAD_ASYNC_TO_LDS_B128 (ASYNCcnt-tracked): signature per clang is
// (v4i addrspace(1)* gsrc, v4i addrspace(3)* ldsdst, imm offset, imm cpol).

#define BATCH 4
#define NPTS  4096
#define KNN   20
#define EPSV  1e-5f
#define SLOPE 0.2f

typedef __attribute__((ext_vector_type(16))) __bf16 v16bf;
typedef __attribute__((ext_vector_type(8)))  float  v8f;
typedef __attribute__((ext_vector_type(4)))  int    v4i;

#if __has_builtin(__builtin_amdgcn_global_load_async_to_lds_b128) && \
    __has_builtin(__builtin_amdgcn_s_wait_asynccnt)
#define HAVE_ASYNC_LDS 1
typedef __attribute__((address_space(1))) v4i gv4i_t;   // global 16B chunk
typedef __attribute__((address_space(3))) v4i lv4i_t;   // LDS 16B chunk
#endif

__device__ __forceinline__ float lrelu(float x) { return x >= 0.f ? x : SLOPE * x; }

// Bulk global->LDS copy, 16B per lane per issue. Async path goes directly
// memory->LDS via the async-LDS engine (no VGPR round trip), tracked by
// ASYNCcnt; caller must async_wait() before __syncthreads().
// LDS destination must be 16B-aligned (use __align__(16) on the arrays).
__device__ __forceinline__ void stage_to_lds(const float* __restrict__ g,
                                             float* l, int nfloats,
                                             int tid, int nthreads) {
#ifdef HAVE_ASYNC_LDS
  for (int c = tid; c < (nfloats >> 2); c += nthreads) {
    __builtin_amdgcn_global_load_async_to_lds_b128(
        (gv4i_t*)(const void*)(g + c * 4), (lv4i_t*)(void*)(l + c * 4), 0, 0);
  }
#else
  for (int i = tid; i < nfloats; i += nthreads) l[i] = g[i];
#endif
}

__device__ __forceinline__ void async_wait() {
#ifdef HAVE_ASYNC_LDS
  __builtin_amdgcn_s_wait_asynccnt(0);
#endif
}

// Register-resident top-K insertion (descending). Strict '>' keeps the earliest
// index on ties, matching jax.lax.top_k tie-breaking.
#define TOPK_INSERT(best, bidx, val, vidx)                                    \
  do {                                                                        \
    float _v = (val);                                                         \
    if (_v > best[KNN - 1]) {                                                 \
      int _i = (vidx);                                                        \
      _Pragma("unroll") for (int _t = 0; _t < KNN; ++_t) {                    \
        if (_v > best[_t]) {                                                  \
          float _tv = best[_t]; int _ti = bidx[_t];                           \
          best[_t] = _v; bidx[_t] = _i; _v = _tv; _i = _ti;                   \
        }                                                                     \
      }                                                                       \
    }                                                                         \
  } while (0)

// ---------------------------------------------------------------------------
// K0: fold BatchNorm params into scale/shift.
// scsh layout (floats): sc1[64] sh1[64] sc2[64] sh2[64] sc5[128] sh5[128] sc6[256] sh6[256]
// ---------------------------------------------------------------------------
__global__ void prep_bn(const float* g1, const float* b1, const float* m1, const float* v1,
                        const float* g2, const float* b2, const float* m2, const float* v2,
                        const float* g5, const float* b5, const float* m5, const float* v5,
                        const float* g6, const float* b6, const float* m6, const float* v6,
                        float* scsh) {
  int t = threadIdx.x;
  if (t < 64) {
    float s = g1[t] * rsqrtf(v1[t] + EPSV);
    scsh[t] = s; scsh[64 + t] = b1[t] - m1[t] * s;
  } else if (t < 128) {
    int i = t - 64;
    float s = g2[i] * rsqrtf(v2[i] + EPSV);
    scsh[128 + i] = s; scsh[192 + i] = b2[i] - m2[i] * s;
  } else if (t < 256) {
    int i = t - 128;
    float s = g5[i] * rsqrtf(v5[i] + EPSV);
    scsh[256 + i] = s; scsh[384 + i] = b5[i] - m5[i] * s;
  } else {
    int i = t - 256;
    float s = g6[i] * rsqrtf(v6[i] + EPSV);
    scsh[512 + i] = s; scsh[768 + i] = b6[i] - m6[i] * s;
  }
}

// ---------------------------------------------------------------------------
// K1: exact KNN on raw 2-channel points. One batch's x fits in 32KB LDS.
// pd(n,j) = 2*x_n.x_j - ||x_n||^2 - ||x_j||^2 ; top-20 (largest) per row.
// ---------------------------------------------------------------------------
__global__ void knn_c2(const float* __restrict__ x, int* __restrict__ idx1) {
  __shared__ __align__(16) float xs0[NPTS];
  __shared__ __align__(16) float xs1[NPTS];
  const int b = blockIdx.y;
  const float* xb = x + (size_t)b * 2 * NPTS;
  stage_to_lds(xb, xs0, NPTS, threadIdx.x, blockDim.x);
  stage_to_lds(xb + NPTS, xs1, NPTS, threadIdx.x, blockDim.x);
  async_wait();
  __syncthreads();
  const int n = blockIdx.x * blockDim.x + threadIdx.x;
  const float cx = xs0[n], cy = xs1[n];
  const float xxn = cx * cx + cy * cy;
  float best[KNN]; int bidx[KNN];
#pragma unroll
  for (int t = 0; t < KNN; ++t) { best[t] = -__builtin_inff(); bidx[t] = 0; }
  for (int j = 0; j < NPTS; ++j) {
    float px = xs0[j], py = xs1[j];
    float d = 2.f * (cx * px + cy * py) - xxn - (px * px + py * py);
    TOPK_INSERT(best, bidx, d, j);
  }
  int* out = idx1 + ((size_t)b * NPTS + n) * KNN;
#pragma unroll
  for (int t = 0; t < KNN; ++t) out[t] = bidx[t];
}

// ---------------------------------------------------------------------------
// K2: edge-conv1 (W1 64x4) + BN + LReLU + max over K=20 -> x1.
// Writes: xc[b, 0:64, n] (f32, channel-major), x1bf[b, n, 0:64] (bf16 point-major
// for coalesced WMMA fragment loads), xx1[b,n] = ||x1||^2.
// ---------------------------------------------------------------------------
__global__ void conv1_max(const float* __restrict__ x, const int* __restrict__ idx1,
                          const float* __restrict__ W1, const float* __restrict__ scsh,
                          float* __restrict__ xc, __bf16* __restrict__ x1bf,
                          float* __restrict__ xx1) {
  __shared__ __align__(16) float xs0[NPTS];
  __shared__ __align__(16) float xs1[NPTS];
  __shared__ __align__(16) float w[64 * 4];
  __shared__ float sc[64], sh[64];
  const int b = blockIdx.y;
  const float* xb = x + (size_t)b * 2 * NPTS;
  stage_to_lds(xb, xs0, NPTS, threadIdx.x, blockDim.x);
  stage_to_lds(xb + NPTS, xs1, NPTS, threadIdx.x, blockDim.x);
  stage_to_lds(W1, w, 256, threadIdx.x, blockDim.x);
  if (threadIdx.x < 64) { sc[threadIdx.x] = scsh[threadIdx.x]; sh[threadIdx.x] = scsh[64 + threadIdx.x]; }
  async_wait();
  __syncthreads();
  const int n = blockIdx.x * blockDim.x + threadIdx.x;
  const float cx = xs0[n], cy = xs1[n];
  const int* id = idx1 + ((size_t)b * NPTS + n) * KNN;
  float ex[KNN], ey[KNN];
#pragma unroll
  for (int k = 0; k < KNN; ++k) { int j = id[k]; ex[k] = xs0[j] - cx; ey[k] = xs1[j] - cy; }
  float* xcb = xc + (size_t)b * 128 * NPTS;
  __bf16* pbf = x1bf + ((size_t)b * NPTS + n) * 64;
  float acc2 = 0.f;
  for (int o = 0; o < 64; ++o) {
    float w0 = w[o * 4], w1 = w[o * 4 + 1], w2c = w[o * 4 + 2], w3 = w[o * 4 + 3];
    float base = w2c * cx + w3 * cy;
    float s = sc[o], hh = sh[o];
    float mv = -__builtin_inff();
#pragma unroll
    for (int k = 0; k < KNN; ++k) {
      float h = w0 * ex[k] + w1 * ey[k] + base;
      mv = fmaxf(mv, lrelu(h * s + hh));
    }
    xcb[(size_t)o * NPTS + n] = mv;
    pbf[o] = (__bf16)mv;
    acc2 += mv * mv;
  }
  xx1[(size_t)b * NPTS + n] = acc2;
}

// ---------------------------------------------------------------------------
// K3: fused KNN#2. One wave owns 32 rows; sweep 256 column tiles of 16.
// Gram tile via 4x V_WMMA_F32_16X16X32_BF16; pd tile dropped through LDS,
// each lane maintains one row's top-20 in registers. Never materializes the
// 4096x4096 distance matrix.
//
// bf16 16x32 A fragment, lane L: M = L&15; K halves {kb..kb+7, kb+16..kb+23},
// kb = (L<16)?0:8. B fragment mirrored with lane = N column.
// ---------------------------------------------------------------------------
__global__ void __launch_bounds__(32) gram_knn(const __bf16* __restrict__ x1bf,
                                               const float* __restrict__ xx1,
                                               int* __restrict__ idx2) {
  __shared__ float tile[32][16];
  __shared__ float xxs[32];
  const int b = blockIdx.y;
  const int r0 = blockIdx.x * 32;
  const int lane = threadIdx.x;
  const __bf16* xb = x1bf + (size_t)b * NPTS * 64;
  const float* xxb = xx1 + (size_t)b * NPTS;
  xxs[lane] = xxb[r0 + lane];
  const int kb = (lane & 16) ? 8 : 0;
  const int lm = lane & 15;
  const int rb = (lane & 16) ? 8 : 0;

  // A fragments for rows [r0, r0+16) and [r0+16, r0+32), K = 0..31 and 32..63.
  v16bf a00, a01, a10, a11;
  {
    const __bf16* p0 = xb + (size_t)(r0 + lm) * 64;
    const __bf16* p1 = xb + (size_t)(r0 + 16 + lm) * 64;
#pragma unroll
    for (int h = 0; h < 8; ++h) {
      a00[h] = p0[kb + h];        a00[8 + h] = p0[kb + 16 + h];
      a01[h] = p0[32 + kb + h];   a01[8 + h] = p0[32 + kb + 16 + h];
      a10[h] = p1[kb + h];        a10[8 + h] = p1[kb + 16 + h];
      a11[h] = p1[32 + kb + h];   a11[8 + h] = p1[32 + kb + 16 + h];
    }
  }

  float best[KNN]; int bidx[KNN];
#pragma unroll
  for (int t = 0; t < KNN; ++t) { best[t] = -__builtin_inff(); bidx[t] = 0; }

  for (int c0 = 0; c0 < NPTS; c0 += 16) {
    v16bf bf0, bf1;
    const __bf16* pc = xb + (size_t)(c0 + lm) * 64;
#pragma unroll
    for (int h = 0; h < 8; ++h) {
      bf0[h] = pc[kb + h];       bf0[8 + h] = pc[kb + 16 + h];
      bf1[h] = pc[32 + kb + h];  bf1[8 + h] = pc[32 + kb + 16 + h];
    }
    v8f acc0 = {}, acc1 = {};
    acc0 = __builtin_amdgcn_wmma_f32_16x16x32_bf16(false, a00, false, bf0, (short)0, acc0, false, false);
    acc0 = __builtin_amdgcn_wmma_f32_16x16x32_bf16(false, a01, false, bf1, (short)0, acc0, false, false);
    acc1 = __builtin_amdgcn_wmma_f32_16x16x32_bf16(false, a10, false, bf0, (short)0, acc1, false, false);
    acc1 = __builtin_amdgcn_wmma_f32_16x16x32_bf16(false, a11, false, bf1, (short)0, acc1, false, false);

    float xxc = xxb[c0 + lm];
#pragma unroll
    for (int v = 0; v < 8; ++v) {
      int rl0 = v + rb, rl1 = 16 + v + rb;
      tile[rl0][lm] = 2.f * acc0[v] - xxs[rl0] - xxc;
      tile[rl1][lm] = 2.f * acc1[v] - xxs[rl1] - xxc;
    }
    __syncthreads();
#pragma unroll
    for (int j = 0; j < 16; ++j) TOPK_INSERT(best, bidx, tile[lane][j], c0 + j);
    __syncthreads();
  }
  int* out = idx2 + ((size_t)b * NPTS + r0 + lane) * KNN;
#pragma unroll
  for (int t = 0; t < KNN; ++t) out[t] = bidx[t];
}

// ---------------------------------------------------------------------------
// K4: edge-conv2 (W2 64x128 on gathered features) + BN + LReLU + max over K.
// W2 (32KB) staged via async-LDS engine; 4 points' center/neighbor features
// gathered into LDS (~54KB total of the WGP's 320KB). Writes xc[b, 64:128, n].
// ---------------------------------------------------------------------------
__global__ void conv2_max(float* __restrict__ xc, const int* __restrict__ idx2,
                          const float* __restrict__ W2, const float* __restrict__ scsh) {
  __shared__ __align__(16) float w2s[64 * 128];
  __shared__ float ctr[4][64];
  __shared__ float nbr[4][KNN][64];
  const int b = blockIdx.y;
  const int n0 = blockIdx.x * 4;
  const int tid = threadIdx.y * 64 + threadIdx.x;
  const float* xcb = xc + (size_t)b * 128 * NPTS;
  stage_to_lds(W2, w2s, 64 * 128, tid, 256);
  for (int i = tid; i < 4 * 64; i += 256) {
    int p = i >> 6, c = i & 63;
    ctr[p][c] = xcb[(size_t)c * NPTS + n0 + p];
  }
  for (int i = tid; i < 4 * KNN * 64; i += 256) {
    int c = i & 63; int rest = i >> 6; int k = rest % KNN; int p = rest / KNN;
    int j = idx2[((size_t)b * NPTS + n0 + p) * KNN + k];
    nbr[p][k][c] = xcb[(size_t)c * NPTS + j];
  }
  async_wait();
  __syncthreads();
  const int o = threadIdx.x, p = threadIdx.y;
  const float s = scsh[128 + o], hh = scsh[192 + o];
  const float* wr = &w2s[o * 128];
  float mv = -__builtin_inff();
  for (int k = 0; k < KNN; ++k) {
    float acc = 0.f;
#pragma unroll
    for (int c = 0; c < 64; ++c) {
      float ct = ctr[p][c];
      acc += (nbr[p][k][c] - ct) * wr[c] + ct * wr[64 + c];
    }
    mv = fmaxf(mv, lrelu(acc * s + hh));
  }
  xc[(size_t)b * 128 * NPTS + (size_t)(64 + o) * NPTS + n0 + p] = mv;
}

// ---------------------------------------------------------------------------
// K5: conv5 = W5(128x128) @ xc(128 x B*N), WMMA bf16, BN+LReLU epilogue -> h5.
// 8 waves per block; wave w computes output rows [16w, 16w+16) for a 16-col tile.
// ---------------------------------------------------------------------------
__global__ void __launch_bounds__(256) conv5_wmma(const float* __restrict__ xc,
                                                  const float* __restrict__ W5,
                                                  const float* __restrict__ scsh,
                                                  float* __restrict__ h5) {
  const int lane = threadIdx.x & 31;
  const int wave = threadIdx.x >> 5;
  const int gcol = blockIdx.x * 16 + (lane & 15);
  const int b = gcol >> 12;
  const int n = gcol & (NPTS - 1);
  const int m0 = wave * 16;
  const int kb = (lane & 16) ? 8 : 0;
  const float* colp = xc + (size_t)b * 128 * NPTS + n;            // [c] stride NPTS
  const float* arow = W5 + (size_t)(m0 + (lane & 15)) * 128;      // row-major weights
  v8f acc = {};
#pragma unroll
  for (int ks = 0; ks < 4; ++ks) {
    const int k0 = ks * 32;
    v16bf af, bfr;
#pragma unroll
    for (int h = 0; h < 8; ++h) {
      af[h]      = (__bf16)arow[k0 + kb + h];
      af[8 + h]  = (__bf16)arow[k0 + kb + 16 + h];
      bfr[h]     = (__bf16)colp[(size_t)(k0 + kb + h) * NPTS];
      bfr[8 + h] = (__bf16)colp[(size_t)(k0 + kb + 16 + h) * NPTS];
    }
    acc = __builtin_amdgcn_wmma_f32_16x16x32_bf16(false, af, false, bfr, (short)0, acc, false, false);
  }
  const int rb = (lane & 16) ? 8 : 0;
#pragma unroll
  for (int v = 0; v < 8; ++v) {
    int o = m0 + v + rb;
    float val = lrelu(acc[v] * scsh[256 + o] + scsh[384 + o]);
    h5[(size_t)b * 128 * NPTS + (size_t)o * NPTS + n] = val;
  }
}

// ---------------------------------------------------------------------------
// K6: gmax[b,o] = max_n h5[b,o,n].
// ---------------------------------------------------------------------------
__global__ void gmax_reduce(const float* __restrict__ h5, float* __restrict__ gmax) {
  __shared__ float red[256];
  const float* p = h5 + (size_t)blockIdx.x * NPTS;
  float mv = -__builtin_inff();
  for (int i = threadIdx.x; i < NPTS; i += 256) mv = fmaxf(mv, p[i]);
  red[threadIdx.x] = mv;
  __syncthreads();
  for (int s = 128; s > 0; s >>= 1) {
    if (threadIdx.x < s) red[threadIdx.x] = fmaxf(red[threadIdx.x], red[threadIdx.x + s]);
    __syncthreads();
  }
  if (threadIdx.x == 0) gmax[blockIdx.x] = red[0];
}

// ---------------------------------------------------------------------------
// K7: conv6 = W6(256x256) @ cat([rep(gmax); xc]), WMMA bf16, BN+LReLU -> h6.
// B-operand synthesizes the concat: c<128 -> gmax broadcast, c>=128 -> xc.
// 16 waves per block, one 16-row M-tile each; K=256 in 8 WMMA steps.
// ---------------------------------------------------------------------------
__global__ void __launch_bounds__(512) conv6_wmma(const float* __restrict__ xc,
                                                  const float* __restrict__ gmax,
                                                  const float* __restrict__ W6,
                                                  const float* __restrict__ scsh,
                                                  float* __restrict__ h6) {
  const int lane = threadIdx.x & 31;
  const int wave = threadIdx.x >> 5;  // 0..15
  const int gcol = blockIdx.x * 16 + (lane & 15);
  const int b = gcol >> 12;
  const int n = gcol & (NPTS - 1);
  const int m0 = wave * 16;
  const int kb = (lane & 16) ? 8 : 0;
  const float* colp = xc + (size_t)b * 128 * NPTS + n;
  const float* gm = gmax + b * 128;
  const float* arow = W6 + (size_t)(m0 + (lane & 15)) * 256;
  v8f acc = {};
#pragma unroll
  for (int ks = 0; ks < 8; ++ks) {
    const int k0 = ks * 32;
    if (ks < 7) __builtin_prefetch(&arow[k0 + 32], 0, 1);  // global_prefetch_b8
    v16bf af, bfr;
#pragma unroll
    for (int h = 0; h < 8; ++h) {
      const int c0 = k0 + kb + h, c1 = k0 + kb + 16 + h;
      af[h]     = (__bf16)arow[c0];
      af[8 + h] = (__bf16)arow[c1];
      float v0 = (c0 < 128) ? gm[c0] : colp[(size_t)(c0 - 128) * NPTS];
      float v1 = (c1 < 128) ? gm[c1] : colp[(size_t)(c1 - 128) * NPTS];
      bfr[h] = (__bf16)v0;
      bfr[8 + h] = (__bf16)v1;
    }
    acc = __builtin_amdgcn_wmma_f32_16x16x32_bf16(false, af, false, bfr, (short)0, acc, false, false);
  }
  const int rb = (lane & 16) ? 8 : 0;
#pragma unroll
  for (int v = 0; v < 8; ++v) {
    int o = m0 + v + rb;
    float val = lrelu(acc[v] * scsh[512 + o] + scsh[768 + o]);
    h6[(size_t)b * 256 * NPTS + (size_t)o * NPTS + n] = val;
  }
}

// ---------------------------------------------------------------------------
// K8: out[b,n] = sum_o W9[o] * h6[b,o,n].
// ---------------------------------------------------------------------------
__global__ void final_dot(const float* __restrict__ h6, const float* __restrict__ W9,
                          float* __restrict__ out) {
  __shared__ __align__(16) float w9s[256];
  stage_to_lds(W9, w9s, 256, threadIdx.x, 256);
  async_wait();
  __syncthreads();
  const int g = blockIdx.x * 256 + threadIdx.x;
  const int b = g >> 12, n = g & (NPTS - 1);
  const float* p = h6 + (size_t)b * 256 * NPTS + n;
  float acc = 0.f;
#pragma unroll 8
  for (int o = 0; o < 256; ++o) acc += w9s[o] * p[(size_t)o * NPTS];
  out[g] = acc;
}

// ---------------------------------------------------------------------------
extern "C" void kernel_launch(void* const* d_in, const int* in_sizes, int n_in,
                              void* d_out, int out_size, void* d_ws, size_t ws_size,
                              hipStream_t stream) {
  (void)in_sizes; (void)n_in; (void)out_size; (void)ws_size;
  const float* x  = (const float*)d_in[0];
  const float* W1 = (const float*)d_in[1];
  const float* g1 = (const float*)d_in[2];
  const float* b1 = (const float*)d_in[3];
  const float* m1 = (const float*)d_in[4];
  const float* v1 = (const float*)d_in[5];
  const float* W2 = (const float*)d_in[6];
  const float* g2 = (const float*)d_in[7];
  const float* b2 = (const float*)d_in[8];
  const float* m2 = (const float*)d_in[9];
  const float* v2 = (const float*)d_in[10];
  const float* W5 = (const float*)d_in[11];
  const float* g5 = (const float*)d_in[12];
  const float* b5 = (const float*)d_in[13];
  const float* m5 = (const float*)d_in[14];
  const float* v5 = (const float*)d_in[15];
  const float* W6 = (const float*)d_in[16];
  const float* g6 = (const float*)d_in[17];
  const float* b6 = (const float*)d_in[18];
  const float* m6 = (const float*)d_in[19];
  const float* v6 = (const float*)d_in[20];
  const float* W9 = (const float*)d_in[21];
  float* out = (float*)d_out;

  // Workspace carve-out (~38 MB), 256B aligned.
  char* ws = (char*)d_ws;
  size_t off = 0;
  auto carve = [&](size_t bytes) {
    void* p = ws + off;
    off = (off + bytes + 255) & ~(size_t)255;
    return p;
  };
  int*    idx1 = (int*)carve((size_t)BATCH * NPTS * KNN * sizeof(int));
  int*    idx2 = (int*)carve((size_t)BATCH * NPTS * KNN * sizeof(int));
  float*  xc   = (float*)carve((size_t)BATCH * 128 * NPTS * sizeof(float));
  __bf16* x1bf = (__bf16*)carve((size_t)BATCH * NPTS * 64 * sizeof(__bf16));
  float*  xx1  = (float*)carve((size_t)BATCH * NPTS * sizeof(float));
  float*  h5   = (float*)carve((size_t)BATCH * 128 * NPTS * sizeof(float));
  float*  gmax = (float*)carve((size_t)BATCH * 128 * sizeof(float));
  float*  h6   = (float*)carve((size_t)BATCH * 256 * NPTS * sizeof(float));
  float*  scsh = (float*)carve(1024 * sizeof(float));

  prep_bn<<<1, 512, 0, stream>>>(g1, b1, m1, v1, g2, b2, m2, v2,
                                 g5, b5, m5, v5, g6, b6, m6, v6, scsh);
  knn_c2<<<dim3(NPTS / 256, BATCH), 256, 0, stream>>>(x, idx1);
  conv1_max<<<dim3(NPTS / 256, BATCH), 256, 0, stream>>>(x, idx1, W1, scsh, xc, x1bf, xx1);
  gram_knn<<<dim3(NPTS / 32, BATCH), 32, 0, stream>>>(x1bf, xx1, idx2);
  conv2_max<<<dim3(NPTS / 4, BATCH), dim3(64, 4), 0, stream>>>(xc, idx2, W2, scsh);
  conv5_wmma<<<(BATCH * NPTS) / 16, 256, 0, stream>>>(xc, W5, scsh, h5);
  gmax_reduce<<<BATCH * 128, 256, 0, stream>>>(h5, gmax);
  conv6_wmma<<<(BATCH * NPTS) / 16, 512, 0, stream>>>(xc, gmax, W6, scsh, h6);
  final_dot<<<(BATCH * NPTS) / 256, 256, 0, stream>>>(h6, W9, out);
}
